// KKAN_Convolutional_Network_56126632624339
// MI455X (gfx1250) — compile-verified, matrix operand order checked by reading, not run on hardware
//
#include <hip/hip_runtime.h>
#include <math.h>

// ---------------------------------------------------------------------------
// KAN conv network for MI455X (gfx1250, wave32, WMMA).
//  Pass 1 (featgen): per-pixel [8 spline bases | silu | 7 zero-pad] -> f16x16.
//         Computed ONCE per pixel (reference recomputes per patch tap, 9x).
//  Pass 2 (gemm):    block = one image row. Stage rows h-1..h+1 feature
//         vectors in LDS, gather 9 neighbors -> A (16x160 f16), combined
//         (spline*scaler | base_w) weights -> B (160x16 f16), 5 k-steps x
//         2 M-tiles of v_wmma_f32_16x16x32_f16. All fragment LDS traffic is
//         16-byte (ds_load_b128) because K is padded to 16 per neighbor.
//  Pass 3 (restore): Conv2d(16,1,3,pad=1) on f16 features, memory bound.
// ---------------------------------------------------------------------------

typedef __attribute__((ext_vector_type(16))) _Float16 v16h;
typedef __attribute__((ext_vector_type(8)))  float    v8f;

#define GRID_SIZE    5
#define SPLINE_ORDER 3
#define NC  16           // output channels of KAN stage
#define P   9            // 3x3 patch taps
#define G   8            // spline coefficients per input
#define FV  16           // padded per-pixel feature vector
#define KT  160          // padded GEMM K: 10 blocks of 16 (9 taps + 1 zero)
#define IMH 256
#define IMW 256

__device__ __forceinline__ float siluf(float x) {
    return x * (1.0f / (1.0f + __expf(-x)));
}

// Cox-de Boor, order 3, uniform knots knot[i] = (i-3)*h - 1, h = 2/5.
__device__ __forceinline__ void bspline8(float x, float bs[G]) {
    const float h = 2.0f / (float)GRID_SIZE;
    float knot[12];
    #pragma unroll
    for (int i = 0; i < 12; ++i) knot[i] = (float)(i - SPLINE_ORDER) * h - 1.0f;
    float b0[11];
    #pragma unroll
    for (int j = 0; j < 11; ++j)
        b0[j] = (x >= knot[j] && x < knot[j + 1]) ? 1.0f : 0.0f;
    const float inv1 = 1.0f / h, inv2 = 1.0f / (2.0f * h), inv3 = 1.0f / (3.0f * h);
    float b1[10];
    #pragma unroll
    for (int j = 0; j < 10; ++j)
        b1[j] = (x - knot[j]) * inv1 * b0[j] + (knot[j + 2] - x) * inv1 * b0[j + 1];
    float b2[9];
    #pragma unroll
    for (int j = 0; j < 9; ++j)
        b2[j] = (x - knot[j]) * inv2 * b1[j] + (knot[j + 3] - x) * inv2 * b1[j + 1];
    #pragma unroll
    for (int j = 0; j < 8; ++j)
        bs[j] = (x - knot[j]) * inv3 * b2[j] + (knot[j + 4] - x) * inv3 * b2[j + 1];
}

union Pack8 { uint4 u; _Float16 h[8]; };

// Copy one 16B chunk (8 f16) into fragment elements [base, base+8).
__device__ __forceinline__ void loadHalf8(v16h& a, int base, const _Float16* src) {
    Pack8 t;
    t.u = *(const uint4*)src;
    #pragma unroll
    for (int e = 0; e < 8; ++e) a[base + e] = t.h[e];
}

// ---------------- Pass 1: per-pixel features ------------------------------
__global__ __launch_bounds__(256)
void kan_featgen_kernel(const float* __restrict__ x,
                        _Float16* __restrict__ featbuf)   // (N, 16) f16
{
    const int n = blockIdx.x * 256 + threadIdx.x;
    const float v = x[n];
    float bs[G];
    bspline8(v, bs);
    Pack8 p0, p1;
    #pragma unroll
    for (int e = 0; e < 8; ++e) p0.h[e] = (_Float16)bs[e];
    #pragma unroll
    for (int e = 0; e < 8; ++e) p1.h[e] = (_Float16)0.0f;
    p1.h[0] = (_Float16)siluf(v);
    uint4* dst = (uint4*)(featbuf + (size_t)n * FV);
    dst[0] = p0.u;
    dst[1] = p1.u;
}

// ---------------- Pass 2: neighbor-gather GEMM via WMMA -------------------
__global__ __launch_bounds__(256)
void kan_gemm_wmma_kernel(const _Float16* __restrict__ featbuf,
                          const float* __restrict__ base_w,        // (16,9)
                          const float* __restrict__ spline_w,      // (16,9,8)
                          const float* __restrict__ spline_scaler, // (16,9)
                          _Float16* __restrict__ feat)             // (N,16) f16
{
    __shared__ __align__(16) _Float16 sWt[NC][KT];            // 5 KB
    __shared__ __align__(16) _Float16 sF[3][IMW + 2][FV];     // 24.8 KB

    const int tid  = threadIdx.x;
    const int bIdx = blockIdx.x;          // one image row per block
    const int b = bIdx >> 8;
    const int h = bIdx & 255;

    // ---- stage combined weights, k = p*16 + f (f<8: spline, f==8: base)
    for (int idx = tid; idx < NC * KT; idx += 256) {
        const int c = idx / KT, k = idx % KT;
        const int p = k >> 4, f = k & 15;
        float wv = 0.0f;
        if (p < P) {
            if (f < G)       wv = spline_w[(c * P + p) * G + f] * spline_scaler[c * P + p];
            else if (f == G) wv = base_w[c * P + p];
        }
        sWt[c][k] = (_Float16)wv;
    }

    // feature vector of a padded zero input (constant-folded b-splines of 0)
    float zb[G];
    bspline8(0.0f, zb);
    Pack8 zf;
    #pragma unroll
    for (int e = 0; e < 8; ++e) zf.h[e] = (_Float16)zb[e];
    const uint4 zero4 = make_uint4(0u, 0u, 0u, 0u);

    // ---- stage rows h-1..h+1 (258 pixels each, zero-pad columns)
    for (int chunk = tid; chunk < 3 * (IMW + 2) * 2; chunk += 256) {
        const int half = chunk & 1, vec = chunk >> 1;
        const int r = vec / (IMW + 2), cc = vec % (IMW + 2);
        const int hh = h - 1 + r, ww = cc - 1;
        uint4 val;
        if (hh >= 0 && hh < IMH && ww >= 0 && ww < IMW) {
            const size_t pix = ((size_t)b << 16) + ((size_t)hh << 8) + (size_t)ww;
            val = *(const uint4*)(featbuf + pix * FV + half * 8);
        } else {
            val = half ? zero4 : zf.u;    // features of x = 0 (silu(0)=0)
        }
        *(uint4*)(&sF[r][cc][half * 8]) = val;
    }
    __syncthreads();

    const int lane = tid & 31;
    const int wave = tid >> 5;
    const int col  = lane & 15;           // D column / B column
    const int hi   = lane >> 4;           // lane-half -> K offset of 8
    const int row  = lane & 15;           // A row within tile
    const int w0   = wave * 32;
    const int wA   = w0 + row;            // tile 0 output pixel (column in row)
    const int wB   = w0 + 16 + row;       // tile 1

    v8f acc0 = {}, acc1 = {};
    #pragma unroll
    for (int s = 0; s < 5; ++s) {
        const int pA = 2 * s, pB = 2 * s + 1;
        v16h a0, a1, bf;
        // B fragment: two 16B chunks of this column's weights
        loadHalf8(bf, 0, &sWt[col][32 * s + 8 * hi]);
        loadHalf8(bf, 8, &sWt[col][32 * s + 16 + 8 * hi]);
        // A fragment halves: neighbor pA (k 32s..) and pB (k 32s+16..)
        {
            const int iA = pA / 3, jA = pA % 3;   // LDS col = w + j (w+j-1 +1)
            loadHalf8(a0, 0, &sF[iA][wA + jA][8 * hi]);
            loadHalf8(a1, 0, &sF[iA][wB + jA][8 * hi]);
        }
        if (pB < P) {
            const int iB = pB / 3, jB = pB % 3;
            loadHalf8(a0, 8, &sF[iB][wA + jB][8 * hi]);
            loadHalf8(a1, 8, &sF[iB][wB + jB][8 * hi]);
        } else {
            #pragma unroll
            for (int e = 8; e < 16; ++e) { a0[e] = (_Float16)0.0f; a1[e] = (_Float16)0.0f; }
        }
        acc0 = __builtin_amdgcn_wmma_f32_16x16x32_f16(
                   false, a0, false, bf, (short)0, acc0, false, false);
        acc1 = __builtin_amdgcn_wmma_f32_16x16x32_f16(
                   false, a1, false, bf, (short)0, acc1, false, false);
    }

    // ---- store D (f32 16x16 layout: VGPR r -> M = r + 8*hi, N = col)
    const size_t rowBase = ((size_t)b << 16) + ((size_t)h << 8);
    #pragma unroll
    for (int r = 0; r < 8; ++r) {
        const int m = r + 8 * hi;
        feat[(rowBase + (size_t)(w0 + m)) * NC + col]      = (_Float16)acc0[r];
        feat[(rowBase + (size_t)(w0 + 16 + m)) * NC + col] = (_Float16)acc1[r];
    }
}

// ---------------- Pass 3: restore Conv2d(16,1,3,pad=1) --------------------
__global__ __launch_bounds__(256)
void kan_restore_kernel(const _Float16* __restrict__ feat,   // (N,16) f16
                        const float* __restrict__ rw,        // (1,16,3,3)
                        const float* __restrict__ rb,        // (1,)
                        float* __restrict__ y, int N)
{
    __shared__ float sw[NC * 9];
    __shared__ float sb;
    const int tid = threadIdx.x;
    if (tid < NC * 9) sw[tid] = rw[tid];
    if (tid == 0) sb = rb[0];
    __syncthreads();

    const int n = blockIdx.x * 256 + tid;
    if (n >= N) return;
    const int b = n >> 16;
    const int h = (n >> 8) & 255;
    const int w = n & 255;

    float acc = sb;
    #pragma unroll
    for (int i = 0; i < 3; ++i) {
        const int hh = h + i - 1;
        if (hh < 0 || hh >= IMH) continue;
        #pragma unroll
        for (int j = 0; j < 3; ++j) {
            const int ww = w + j - 1;
            if (ww < 0 || ww >= IMW) continue;
            const size_t pix = ((size_t)b << 16) + ((size_t)hh << 8) + (size_t)ww;
            const _Float16* fp = feat + pix * NC;
            Pack8 t0, t1;
            t0.u = *(const uint4*)fp;
            t1.u = *(const uint4*)(fp + 8);
            const int t = i * 3 + j;
            #pragma unroll
            for (int c = 0; c < 8; ++c) acc += (float)t0.h[c] * sw[c * 9 + t];
            #pragma unroll
            for (int c = 0; c < 8; ++c) acc += (float)t1.h[c] * sw[(8 + c) * 9 + t];
        }
    }
    y[n] = acc;
}

extern "C" void kernel_launch(void* const* d_in, const int* in_sizes, int n_in,
                              void* d_out, int out_size, void* d_ws, size_t ws_size,
                              hipStream_t stream) {
    const float* x             = (const float*)d_in[0];
    const float* base_w        = (const float*)d_in[1];
    const float* spline_w      = (const float*)d_in[2];
    const float* spline_scaler = (const float*)d_in[3];
    const float* restore_w     = (const float*)d_in[4];
    const float* restore_b     = (const float*)d_in[5];
    float* y = (float*)d_out;

    const int N = in_sizes[0];             // B*H*W = 2,097,152
    (void)n_in; (void)out_size; (void)ws_size;

    _Float16* featbuf = (_Float16*)d_ws;                                   // N*16 f16 (64 MB)
    _Float16* feat    = (_Float16*)((char*)d_ws +
                            (size_t)N * FV * sizeof(_Float16));            // N*16 f16 (64 MB)

    kan_featgen_kernel<<<N / 256, 256, 0, stream>>>(x, featbuf);
    kan_gemm_wmma_kernel<<<N / IMW, 256, 0, stream>>>(
        featbuf, base_w, spline_w, spline_scaler, feat);
    kan_restore_kernel<<<(N + 255) / 256, 256, 0, stream>>>(
        feat, restore_w, restore_b, y, N);
}
